// LocalEncoderBlock_15848429322821
// MI455X (gfx1250) — matrix operand order repaired and checked by reading, not compiled, not run on hardware
//
#include <hip/hip_runtime.h>
#include <hip/hip_bf16.h>
#include <math.h>

typedef _Float16 half_t;
typedef __attribute__((ext_vector_type(16))) _Float16 v16h;
typedef __attribute__((ext_vector_type(8)))  float    v8f;
typedef __attribute__((ext_vector_type(4)))  unsigned int u32x4;
typedef __attribute__((ext_vector_type(8)))  int          i32x8;
typedef __attribute__((ext_vector_type(4)))  int          i32x4;

#define AS1 __attribute__((address_space(1)))
#define AS3 __attribute__((address_space(3)))

#if __has_builtin(__builtin_amdgcn_tensor_load_to_lds)
#define HAVE_TDM 1
#else
#define HAVE_TDM 0
#endif

#if __has_builtin(__builtin_amdgcn_global_load_async_to_lds_b128)
#define HAVE_ASYNC 1
#else
#define HAVE_ASYNC 0
#endif

union V16H { v16h v; half_t h[16]; uint4 q[2]; };

__device__ __forceinline__ v8f wmma_f16(v16h a, v16h b, v8f c) {
  // D = A(16x32 f16) * B(32x16 f16) + C(16x16 f32)
  return __builtin_amdgcn_wmma_f32_16x16x32_f16(false, a, false, b, (short)0, c, false, false);
}

__device__ __forceinline__ void tdm_wait() {
#if __has_builtin(__builtin_amdgcn_s_wait_tensorcnt)
  __builtin_amdgcn_s_wait_tensorcnt((short)0);
#else
  asm volatile("s_wait_tensorcnt 0x0" ::: "memory");
#endif
}

__device__ __forceinline__ void async_wait() {
#if __has_builtin(__builtin_amdgcn_s_wait_asynccnt)
  __builtin_amdgcn_s_wait_asynccnt((short)0);
#else
  asm volatile("s_wait_asynccnt 0x0" ::: "memory");
#endif
}

// Issue a TDM 2D tile load: tile_x (contiguous, elements) x tile_y rows,
// row stride = tensor_w elements, element = 2 bytes, LDS padded by 4 dwords
// every 16 dwords (matches a [rows][40] half_t LDS layout for tile_x == 32).
__device__ __forceinline__ void tdm_load_tile_2d(const half_t* gsrc, unsigned lds_byte_addr,
                                                 unsigned tile_x, unsigned tile_y,
                                                 unsigned tensor_w, unsigned tensor_h) {
#if HAVE_TDM
  unsigned long long ga = (unsigned long long)(size_t)gsrc;
  u32x4 g0;
  g0.x = 1u;                                  // count=1, user descriptor
  g0.y = lds_byte_addr;                       // lds_addr
  g0.z = (unsigned)(ga & 0xffffffffu);        // global_addr[31:0]
  g0.w = (unsigned)(ga >> 32) | (2u << 30);   // global_addr[56:32] | type=2 (image)
  i32x8 g1;
  g1[0] = (1 << 16)     // data_size = 2 bytes
        | (1 << 20)     // pad_enable
        | (3 << 22)     // pad_interval: 16 dwords
        | (3 << 25);    // pad_amount: 4 dwords
  g1[1] = (int)((tensor_w & 0xffffu) << 16);                        // tensor_dim0[15:0]
  g1[2] = (int)((tensor_w >> 16) | ((tensor_h & 0xffffu) << 16));   // td0 hi | td1 lo
  g1[3] = (int)((tensor_h >> 16) | ((tile_x & 0xffffu) << 16));     // td1 hi | tile_dim0
  g1[4] = (int)(tile_y & 0xffffu);                                  // tile_dim1 (tile_dim2=0)
  g1[5] = (int)tensor_w;                                            // tensor_dim0_stride lo
  g1[6] = 0;
  g1[7] = 0;
  i32x4 z4 = {0, 0, 0, 0};
#if __clang_major__ >= 23
  i32x8 z8 = {0, 0, 0, 0, 0, 0, 0, 0};
  __builtin_amdgcn_tensor_load_to_lds(g0, g1, z4, z4, z8, 0);
#else
  __builtin_amdgcn_tensor_load_to_lds(g0, g1, z4, z4, 0);
#endif
#else
  (void)gsrc; (void)lds_byte_addr; (void)tile_x; (void)tile_y; (void)tensor_w; (void)tensor_h;
#endif
}

#define BATCH 4
#define SEQ   4096
#define CH    512
#define FF    2048
#define NH    8
#define HD    64
#define WIN   128
#define NWIN  32
#define RTOP  1024
#define MROWS (BATCH*SEQ)

// ---------------------------------------------------------------- cvt f32->f16
__global__ __launch_bounds__(256) void k_cvt(const float* __restrict__ in, half_t* __restrict__ out, int n) {
  int i = blockIdx.x * 256 + threadIdx.x;
  if (i < n) out[i] = (half_t)in[i];
}

// -------------------------------------------- head-averaged K projection weight
__global__ __launch_bounds__(256) void k_wkavg(const float* __restrict__ Wqkv, const float* __restrict__ bqkv,
                                               half_t* __restrict__ wk, float* __restrict__ bk) {
  int i = blockIdx.x * 256 + threadIdx.x;   // 512*64
  int c = i >> 6, d = i & 63;
  float s = 0.f;
#pragma unroll
  for (int h = 0; h < NH; ++h) s += Wqkv[c * (3 * CH) + CH + h * HD + d];
  wk[c * HD + d] = (half_t)(s * 0.125f);
  if (i < HD) {
    float t = 0.f;
#pragma unroll
    for (int h = 0; h < NH; ++h) t += bqkv[CH + h * HD + i];
    bk[i] = t * 0.125f;
  }
}

// ---------------------------------------------------------------- layernorm (wave per row)
__global__ __launch_bounds__(256) void k_ln(const float* __restrict__ x, const float* __restrict__ g,
                                            const float* __restrict__ b, half_t* __restrict__ out) {
  int row = blockIdx.x * 8 + (threadIdx.x >> 5);
  int lane = threadIdx.x & 31;
  const float* xr = x + (size_t)row * CH;
  float v[16];
  float s = 0.f;
#pragma unroll
  for (int i = 0; i < 16; ++i) { v[i] = xr[lane + i * 32]; s += v[i]; }
#pragma unroll
  for (int m = 16; m >= 1; m >>= 1) s += __shfl_xor(s, m, 32);
  float mu = s * (1.f / CH);
  float vs = 0.f;
#pragma unroll
  for (int i = 0; i < 16; ++i) { float d = v[i] - mu; vs += d * d; }
#pragma unroll
  for (int m = 16; m >= 1; m >>= 1) vs += __shfl_xor(vs, m, 32);
  float rs = rsqrtf(vs * (1.f / CH) + 1e-5f);
  half_t* orow = out + (size_t)row * CH;
#pragma unroll
  for (int i = 0; i < 16; ++i) {
    int c = lane + i * 32;
    orow[c] = (half_t)((v[i] - mu) * rs * g[c] + b[c]);
  }
}

// ---------------------------------------------------------------- generic WMMA GEMM
// C[M,N] = A[M,K](f16) @ B[K,N](f16) + bias (+resid) (gelu?) -> f16 or f32
// block: 256 threads (8 waves); tile 128x64; each wave: 16 rows x 64 cols (4 wmma accs)
// A tile staged through the Tensor Data Mover when available (wave 0 issues the
// descriptor, TENSORcnt-waited before the block barrier).
template <bool GELU, bool RESID, bool OUT_HALF>
__global__ __launch_bounds__(256) void k_gemm(const half_t* __restrict__ A, const half_t* __restrict__ Bw,
                                              const float* __restrict__ bias, const float* __restrict__ resid,
                                              void* __restrict__ outp, int M, int K, int Ncols) {
  __shared__ half_t Asl[128][40];   // [row][k]  (k-chunk of 32, 8-half pad == TDM pad 4 dwords/16 dwords)
  __shared__ half_t Bsl[64][40];    // [col][k]  transposed for contiguous B-frags

  int tid = threadIdx.x;
  int wv = tid >> 5, lane = tid & 31;
  int li = lane & 15, hi = lane >> 4;
  int tileM = blockIdx.y * 128;
  int tileN = blockIdx.x * 64;

  v8f acc[4] = {};

  int ar = tid >> 1, ac = (tid & 1) * 16;  // A: 2 threads/row, 16 halves each (fallback path)
  int br = tid >> 3, bc = (tid & 7) * 8;   // B: 8 threads/row, 8 halves each
  int ksteps = K >> 5;
#if HAVE_TDM
  unsigned lds_a = (unsigned)(size_t)(AS3 char*)&Asl[0][0];
#endif

  for (int ks = 0; ks < ksteps; ++ks) {
    int kk = ks << 5;
#if HAVE_TDM
    if (wv == 0)
      tdm_load_tile_2d(A + (size_t)tileM * K + kk, lds_a, 32u, 128u, (unsigned)K, (unsigned)M);
#else
    const half_t* ag = A + (size_t)(tileM + ar) * K + kk + ac;
    *(uint4*)&Asl[ar][ac]     = *(const uint4*)ag;
    *(uint4*)&Asl[ar][ac + 8] = *(const uint4*)(ag + 8);
#endif
    const half_t* bg = Bw + (size_t)(kk + br) * Ncols + tileN + bc;
    V16H bt;
    bt.q[0] = *(const uint4*)bg;
#pragma unroll
    for (int q = 0; q < 8; ++q) Bsl[bc + q][br] = bt.h[q];
    if (ks + 1 < ksteps) __builtin_prefetch(bg + 32 * (size_t)Ncols, 0, 0);
#if HAVE_TDM
    if (wv == 0) tdm_wait();
#endif
    __syncthreads();

    // A fragment: lanes 0-15 rows, k chunks [hi*8,+8) and [16+hi*8,+8)
    V16H af;
    const half_t* ap = &Asl[wv * 16 + li][hi * 8];
    af.q[0] = *(const uint4*)ap;
    af.q[1] = *(const uint4*)(ap + 16);
#pragma unroll
    for (int ct = 0; ct < 4; ++ct) {
      V16H bf;  // B fragment: k chunks [hi*8,+8) and [16+hi*8,+8) (transposed LDS)
      const half_t* bp = &Bsl[ct * 16 + li][hi * 8];
      bf.q[0] = *(const uint4*)bp;
      bf.q[1] = *(const uint4*)(bp + 16);
      acc[ct] = wmma_f16(af.v, bf.v, acc[ct]);
    }
    __syncthreads();
  }

#pragma unroll
  for (int ct = 0; ct < 4; ++ct) {
    int col = tileN + ct * 16 + li;
    float bv = bias[col];
#pragma unroll
    for (int r = 0; r < 8; ++r) {
      int row = tileM + wv * 16 + r + hi * 8;
      float v = acc[ct][r] + bv;
      if (RESID) v += resid[(size_t)row * Ncols + col];
      if (GELU)  v = 0.5f * v * (1.0f + erff(v * 0.70710678118f));
      if (OUT_HALF) ((half_t*)outp)[(size_t)row * Ncols + col] = (half_t)v;
      else          ((float*)outp)[(size_t)row * Ncols + col] = v;
    }
  }
}

// ---------------------------------------------------------------- local windowed flash attention
// one block per (b, head, window); 8 waves, each wave owns 16 query rows
__global__ __launch_bounds__(256) void k_attn(const half_t* __restrict__ qkvh, half_t* __restrict__ aoh) {
  __shared__ half_t ksh[32][72];        // [key][d]
  __shared__ half_t vsh[64][40];        // [d][key]  (transposed)
  __shared__ float  ssc[8][16][33];     // per-wave raw scores 16x32
  __shared__ half_t psc[8][16][40];     // per-wave probabilities 16x32 (f16)
  __shared__ float  mrow[8][16], lrow[8][16], srow[8][16];

  int tid = threadIdx.x;
  int wv = tid >> 5, lane = tid & 31, li = lane & 15, hi = lane >> 4;
  int blk = blockIdx.x;
  int win = blk & 31;
  int hh  = (blk >> 5) & 7;
  int bb  = blk >> 8;

  // q fragments (2 x 16x32 along d), pre-scaled by 1/sqrt(64)
  int qrow = win * WIN + wv * 16 + li;
  const half_t* qbase = qkvh + ((size_t)(bb * SEQ + qrow)) * (3 * CH) + hh * HD;
  V16H qf[2];
#pragma unroll
  for (int ds = 0; ds < 2; ++ds) {
    const half_t* qp = qbase + ds * 32 + hi * 8;
    qf[ds].q[0] = *(const uint4*)qp;
    qf[ds].q[1] = *(const uint4*)(qp + 16);
#pragma unroll
    for (int t = 0; t < 16; ++t) qf[ds].h[t] = (half_t)((float)qf[ds].h[t] * 0.125f);
  }
  if (hi == 0) { mrow[wv][li] = -3.0e38f; lrow[wv][li] = 0.f; }
  v8f acco[4] = {};

  int klr = tid >> 3;           // key-local row 0..31
  int kd0 = (tid & 7) * 8;      // 8 halves of d

  for (int st = 0; st < 12; ++st) {
    int kbase = win * WIN - WIN + st * 32;   // global key index of tile start
    __syncthreads();
    // cooperative K/V tile load (zero-fill out-of-range)
    int kg = kbase + klr;
    uint4 kq = {0, 0, 0, 0}, vq = {0, 0, 0, 0};
    if (kg >= 0 && kg < SEQ) {
      const half_t* kp = qkvh + ((size_t)(bb * SEQ + kg)) * (3 * CH) + CH + hh * HD + kd0;
      kq = *(const uint4*)kp;
      vq = *(const uint4*)(kp + CH);
    }
    *(uint4*)&ksh[klr][kd0] = kq;
    V16H vt; vt.q[0] = vq;
#pragma unroll
    for (int q = 0; q < 8; ++q) vsh[kd0 + q][klr] = vt.h[q];
    __syncthreads();

    // s = q @ k^T : two 16x16 tiles, K=64 in two wmma steps
    v8f sacc[2] = {};
#pragma unroll
    for (int ct = 0; ct < 2; ++ct) {
#pragma unroll
      for (int ds = 0; ds < 2; ++ds) {
        V16H kf;
        const half_t* kp2 = &ksh[ct * 16 + li][ds * 32 + hi * 8];
        kf.q[0] = *(const uint4*)kp2;
        kf.q[1] = *(const uint4*)(kp2 + 16);
        sacc[ct] = wmma_f16(qf[ds].v, kf.v, sacc[ct]);
      }
#pragma unroll
      for (int r = 0; r < 8; ++r) ssc[wv][r + hi * 8][ct * 16 + li] = sacc[ct][r];
    }
    __syncthreads();

    // online softmax update: 2 lanes per row, 16 cols each
    float mo = mrow[wv][li];
    float lmax = -3.0e38f;
#pragma unroll
    for (int c = 0; c < 16; ++c) {
      int col = hi * 16 + c;
      int kgc = kbase + col;
      float sv = (kgc >= 0 && kgc < SEQ) ? ssc[wv][li][col] : -3.0e38f;
      lmax = fmaxf(lmax, sv);
    }
    lmax = fmaxf(lmax, __shfl_xor(lmax, 16, 32));
    float nm = fmaxf(mo, lmax);
    float ls = 0.f;
#pragma unroll
    for (int c = 0; c < 16; ++c) {
      int col = hi * 16 + c;
      int kgc = kbase + col;
      float p = 0.f;
      if (kgc >= 0 && kgc < SEQ) p = __expf(ssc[wv][li][col] - nm);
      psc[wv][li][col] = (half_t)p;
      ls += p;
    }
    ls += __shfl_xor(ls, 16, 32);
    float sc = __expf(mo - nm);
    if (hi == 0) {
      mrow[wv][li] = nm;
      lrow[wv][li] = lrow[wv][li] * sc + ls;
      srow[wv][li] = sc;
    }
    __syncthreads();

    // rescale running output, then out += P @ V
#pragma unroll
    for (int r = 0; r < 8; ++r) {
      float f = srow[wv][r + hi * 8];
#pragma unroll
      for (int ct = 0; ct < 4; ++ct) acco[ct][r] *= f;
    }
    V16H pf;
    const half_t* pp = &psc[wv][li][hi * 8];
    pf.q[0] = *(const uint4*)pp;
    pf.q[1] = *(const uint4*)(pp + 16);
#pragma unroll
    for (int ct = 0; ct < 4; ++ct) {
      V16H vf;
      const half_t* vp = &vsh[ct * 16 + li][hi * 8];
      vf.q[0] = *(const uint4*)vp;
      vf.q[1] = *(const uint4*)(vp + 16);
      acco[ct] = wmma_f16(pf.v, vf.v, acco[ct]);
    }
  }
  __syncthreads();

#pragma unroll
  for (int ct = 0; ct < 4; ++ct) {
#pragma unroll
    for (int r = 0; r < 8; ++r) {
      int row = r + hi * 8;
      float l = lrow[wv][row];
      float o = (l > 0.f) ? acco[ct][r] / l : 0.f;
      int n = win * WIN + wv * 16 + row;
      aoh[((size_t)(bb * SEQ + n)) * CH + hh * HD + ct * 16 + li] = (half_t)o;
    }
  }
}

// ---------------------------------------------------------------- keys normalization (wave per row)
__global__ __launch_bounds__(256) void k_knorm(const float* __restrict__ keys, half_t* __restrict__ knh) {
  int row = blockIdx.x * 8 + (threadIdx.x >> 5);
  int lane = threadIdx.x & 31;
  const float* kr = keys + (size_t)row * HD;
  float a = kr[lane], b = kr[32 + lane];
  float s = a * a + b * b;
#pragma unroll
  for (int m = 16; m >= 1; m >>= 1) s += __shfl_xor(s, m, 32);
  float rn = rsqrtf(s);
  knh[(size_t)row * HD + lane]      = (half_t)(a * rn);
  knh[(size_t)row * HD + 32 + lane] = (half_t)(b * rn);
}

// ---------------------------------------------------------------- scores + row argmax (WMMA)
__global__ __launch_bounds__(256) void k_scores(const half_t* __restrict__ knh,
                                                float* __restrict__ node_max, int* __restrict__ node_idx) {
  __shared__ float ssc[8][16][17];
  __shared__ float rmax[8][16];
  __shared__ int   ridx[8][16];
  int tid = threadIdx.x, wv = tid >> 5, lane = tid & 31, li = lane & 15, hi = lane >> 4;
  int bb = blockIdx.y;
  int i = blockIdx.x * 128 + wv * 16 + li;  // a-row in [0,2048)

  const half_t* abase = knh + ((size_t)(bb * SEQ + 2 * i)) * HD;
  V16H af[2];
#pragma unroll
  for (int ds = 0; ds < 2; ++ds) {
    const half_t* ap = abase + ds * 32 + hi * 8;
    af[ds].q[0] = *(const uint4*)ap;
    af[ds].q[1] = *(const uint4*)(ap + 16);
  }
  if (hi == 0) { rmax[wv][li] = -3.0e38f; ridx[wv][li] = 0; }

  for (int j0 = 0; j0 < 2048; j0 += 16) {
    v8f acc = {};
#pragma unroll
    for (int ds = 0; ds < 2; ++ds) {
      V16H bf;
      const half_t* bp = knh + ((size_t)(bb * SEQ + 2 * (j0 + li) + 1)) * HD + ds * 32 + hi * 8;
      bf.q[0] = *(const uint4*)bp;
      bf.q[1] = *(const uint4*)(bp + 16);
      acc = wmma_f16(af[ds].v, bf.v, acc);
    }
#pragma unroll
    for (int r = 0; r < 8; ++r) ssc[wv][r + hi * 8][li] = acc[r];
    __syncthreads();
    float bmax = -3.0e38f; int bidx = 0;
#pragma unroll
    for (int c = 0; c < 8; ++c) {
      int col = hi * 8 + c;
      float v = ssc[wv][li][col];
      if (v > bmax) { bmax = v; bidx = j0 + col; }
    }
    float om = __shfl_xor(bmax, 16, 32);
    int   oi = __shfl_xor(bidx, 16, 32);
    if (om > bmax || (om == bmax && oi < bidx)) { bmax = om; bidx = oi; }
    if (hi == 0 && bmax > rmax[wv][li]) { rmax[wv][li] = bmax; ridx[wv][li] = bidx; }
    __syncthreads();
  }
  if (hi == 0) {
    node_max[bb * 2048 + i] = rmax[wv][li];
    node_idx[bb * 2048 + i] = ridx[wv][li];
  }
}

// ---------------------------------------------------------------- bitonic sort per batch (ToMe argsort)
__global__ __launch_bounds__(1024) void k_sort(const float* __restrict__ node_max, const int* __restrict__ node_idx,
                                               int* __restrict__ src_idx, int* __restrict__ unm_idx,
                                               int* __restrict__ dst_idx) {
  __shared__ float sv[2048];
  __shared__ int   si[2048];
  __shared__ int   ti[1024];
  int tid = threadIdx.x, bb = blockIdx.x;
#if HAVE_ASYNC
  if (tid < 512)
    __builtin_amdgcn_global_load_async_to_lds_b128(
        (AS1 i32x4*)(node_max + bb * 2048 + tid * 4),
        (AS3 i32x4*)&sv[tid * 4], 0, 0);
  for (int t = tid; t < 2048; t += 1024) si[t] = t;
  async_wait();
#else
  for (int t = tid; t < 2048; t += 1024) { sv[t] = node_max[bb * 2048 + t]; si[t] = t; }
#endif
  __syncthreads();
  // descending by value, stable (tie -> smaller index first)
  for (int k = 2; k <= 2048; k <<= 1) {
    for (int j = k >> 1; j > 0; j >>= 1) {
      for (int t = tid; t < 2048; t += 1024) {
        int p = t ^ j;
        if (p > t) {
          float va = sv[t], vb = sv[p];
          int ia = si[t], ib = si[p];
          bool up = ((t & k) == 0);
          bool prec = (va > vb) || (va == vb && ia < ib);
          if (prec != up) { sv[t] = vb; sv[p] = va; si[t] = ib; si[p] = ia; }
        }
      }
      __syncthreads();
    }
  }
  int s = si[tid];
  src_idx[bb * RTOP + tid] = s;
  dst_idx[bb * RTOP + tid] = node_idx[bb * 2048 + s];
  ti[tid] = si[RTOP + tid];
  __syncthreads();
  // ascending int sort of the unmerged indices
  for (int k = 2; k <= 1024; k <<= 1) {
    for (int j = k >> 1; j > 0; j >>= 1) {
      int p = tid ^ j;
      if (p > tid) {
        int a = ti[tid], b2 = ti[p];
        bool up = ((tid & k) == 0);
        if ((a < b2) != up) { ti[tid] = b2; ti[p] = a; }
      }
      __syncthreads();
    }
  }
  unm_idx[bb * RTOP + tid] = ti[tid];
}

// ---------------------------------------------------------------- token merge
__global__ __launch_bounds__(256) void k_merge_copy(const float* __restrict__ t, const int* __restrict__ unm,
                                                    float* __restrict__ out, int width) {
  int bb = blockIdx.x / 3072;
  int r  = blockIdx.x % 3072;
  int srow = (r < RTOP) ? 2 * unm[bb * RTOP + r] : 2 * (r - RTOP) + 1;
  const float* s = t + ((size_t)bb * SEQ + srow) * width;
  float* d = out + ((size_t)bb * 3072 + r) * width;
  for (int c = threadIdx.x; c < width; c += 256) d[c] = s[c];
}

__global__ __launch_bounds__(256) void k_merge_scatter(const float* __restrict__ t, const int* __restrict__ srci,
                                                       const int* __restrict__ dsti, float* __restrict__ out,
                                                       int width) {
  int bb = blockIdx.x / RTOP;
  int i  = blockIdx.x % RTOP;
  int srow = 2 * srci[bb * RTOP + i];
  int drow = RTOP + dsti[bb * RTOP + i];
  const float* s = t + ((size_t)bb * SEQ + srow) * width;
  float* d = out + ((size_t)bb * 3072 + drow) * width;
  for (int c = threadIdx.x; c < width; c += 256) atomicAdd(&d[c], s[c]);
}

// ================================================================ launcher
extern "C" void kernel_launch(void* const* d_in, const int* in_sizes, int n_in,
                              void* d_out, int out_size, void* d_ws, size_t ws_size,
                              hipStream_t stream) {
  (void)in_sizes; (void)n_in; (void)out_size; (void)ws_size;
  const float* x    = (const float*)d_in[0];
  const float* S    = (const float*)d_in[1];
  const float* g1   = (const float*)d_in[2];
  const float* b1   = (const float*)d_in[3];
  const float* Wqkv = (const float*)d_in[4];
  const float* bqkv = (const float*)d_in[5];
  const float* Wp   = (const float*)d_in[6];
  const float* bp   = (const float*)d_in[7];
  const float* g2   = (const float*)d_in[8];
  const float* b2   = (const float*)d_in[9];
  const float* W1   = (const float*)d_in[10];
  const float* bm1  = (const float*)d_in[11];
  const float* W2   = (const float*)d_in[12];
  const float* bm2  = (const float*)d_in[13];

  char* ws = (char*)d_ws;
  size_t off = 0;
  auto alloc = [&](size_t bytes) -> void* {
    void* p = (void*)(ws + off);
    off += (bytes + 255) & ~(size_t)255;
    return p;
  };

  half_t* wqkvh = (half_t*)alloc((size_t)CH * 3 * CH * 2);
  half_t* wph   = (half_t*)alloc((size_t)CH * CH * 2);
  half_t* w1h   = (half_t*)alloc((size_t)CH * FF * 2);
  half_t* w2h   = (half_t*)alloc((size_t)FF * CH * 2);
  half_t* wkh   = (half_t*)alloc((size_t)CH * HD * 2);
  float*  bk    = (float*)alloc(HD * 4);
  half_t* bufA  = (half_t*)alloc((size_t)MROWS * CH * 2);     // xn1, then ao
  half_t* bufQ  = (half_t*)alloc((size_t)MROWS * 3 * CH * 2); // qkv, then xn2, xn3
  float*  x2    = (float*)alloc((size_t)MROWS * CH * 4);
  half_t* bufH  = (half_t*)alloc((size_t)MROWS * FF * 2);     // h1, then keys(f32)+kn(f16)
  float*  x3    = (float*)alloc((size_t)MROWS * CH * 4);
  float*  nmax  = (float*)alloc(BATCH * 2048 * 4);
  int*    nidx  = (int*)alloc(BATCH * 2048 * 4);
  int*    srci  = (int*)alloc(BATCH * RTOP * 4);
  int*    unmi  = (int*)alloc(BATCH * RTOP * 4);
  int*    dsti  = (int*)alloc(BATCH * RTOP * 4);
  float*  keys  = (float*)bufH;
  half_t* knh   = bufH + (size_t)MROWS * HD * 2;  // keys occupies MROWS*HD floats = MROWS*HD*2 halves

  // weights -> f16 (+ folded head-mean K weight)
  k_cvt<<<(CH * 3 * CH + 255) / 256, 256, 0, stream>>>(Wqkv, wqkvh, CH * 3 * CH);
  k_cvt<<<(CH * CH + 255) / 256, 256, 0, stream>>>(Wp, wph, CH * CH);
  k_cvt<<<(CH * FF + 255) / 256, 256, 0, stream>>>(W1, w1h, CH * FF);
  k_cvt<<<(FF * CH + 255) / 256, 256, 0, stream>>>(W2, w2h, FF * CH);
  k_wkavg<<<(CH * HD + 255) / 256, 256, 0, stream>>>(Wqkv, bqkv, wkh, bk);

  // attn block
  k_ln<<<MROWS / 8, 256, 0, stream>>>(x, g1, b1, bufA);
  k_gemm<false, false, true><<<dim3(3 * CH / 64, MROWS / 128), 256, 0, stream>>>(
      bufA, wqkvh, bqkv, nullptr, bufQ, MROWS, CH, 3 * CH);
  k_attn<<<BATCH * NH * NWIN, 256, 0, stream>>>(bufQ, bufA);
  k_gemm<false, true, false><<<dim3(CH / 64, MROWS / 128), 256, 0, stream>>>(
      bufA, wph, bp, x, x2, MROWS, CH, CH);
  k_ln<<<MROWS / 8, 256, 0, stream>>>(x2, g2, b2, bufQ);
  k_gemm<true, false, true><<<dim3(FF / 64, MROWS / 128), 256, 0, stream>>>(
      bufQ, w1h, bm1, nullptr, bufH, MROWS, CH, FF);
  k_gemm<false, true, false><<<dim3(CH / 64, MROWS / 128), 256, 0, stream>>>(
      bufH, w2h, bm2, x2, x3, MROWS, FF, CH);

  // get_keys + normalize + scores + sort
  k_ln<<<MROWS / 8, 256, 0, stream>>>(x3, g1, b1, bufQ);
  k_gemm<false, false, false><<<dim3(1, MROWS / 128), 256, 0, stream>>>(
      bufQ, wkh, bk, nullptr, keys, MROWS, CH, HD);
  k_knorm<<<MROWS / 8, 256, 0, stream>>>(keys, knh);
  k_scores<<<dim3(16, BATCH), 256, 0, stream>>>(knh, nmax, nidx);
  k_sort<<<BATCH, 1024, 0, stream>>>(nmax, nidx, srci, unmi, dsti);

  // merge(x3), merge(S)
  float* outx = (float*)d_out;
  float* outS = outx + (size_t)BATCH * 3072 * CH;
  k_merge_copy<<<BATCH * 3072, 256, 0, stream>>>(x3, unmi, outx, CH);
  k_merge_scatter<<<BATCH * RTOP, 256, 0, stream>>>(x3, srci, dsti, outx, CH);
  k_merge_copy<<<BATCH * 3072, 256, 0, stream>>>(S, unmi, outS, SEQ);
  k_merge_scatter<<<BATCH * RTOP, 256, 0, stream>>>(S, srci, dsti, outS, SEQ);
}
